// SepConv_15616501088614
// MI455X (gfx1250) — matrix operand order repaired and checked by reading, not compiled, not run on hardware
//
#include <hip/hip_runtime.h>
#include <cstdint>

// ---------------------------------------------------------------------------
// Types for WMMA (CDNA5 / gfx1250, wave32)
// ---------------------------------------------------------------------------
typedef __attribute__((ext_vector_type(16))) _Float16 v16h;
typedef __attribute__((ext_vector_type(8)))  float    v8f;

static __device__ __forceinline__ int imin(int a, int b) { return a < b ? a : b; }
static __device__ __forceinline__ int imax(int a, int b) { return a > b ? a : b; }

// ---------------------------------------------------------------------------
// Pack f32 weights [Cout][K] into the exact WMMA A-fragment layout:
//   dst[((ct*K32 + ks)*32 + lane)*16 + e]  (f16, zero-padded)
// Lane layout (16-bit A 16x32): lane l16 holds row M,
// halves: K = half*16 + e (contiguous, matches B), +32 per ks.
// In-kernel the A fragment is then ONE contiguous 32B global load per lane.
// ---------------------------------------------------------------------------
__global__ void pack_wmma_weights_kernel(const float* __restrict__ w,
                                         _Float16* __restrict__ dst,
                                         int Cout, int K, int K32, int CoT) {
    long long total = (long long)CoT * K32 * 512;  // halves
    long long i = (long long)blockIdx.x * blockDim.x + threadIdx.x;
    if (i >= total) return;
    int e    = (int)(i & 15);
    int lane = (int)((i >> 4) & 31);
    long long rest = i >> 9;                // ct * K32 + ks
    int ks = (int)(rest % K32);
    int ct = (int)(rest / K32);
    int half = lane >> 4, l16 = lane & 15;
    int m  = ct * 16 + l16;
    int kk = ks * 32 + half * 16 + e;
    _Float16 v = (_Float16)0.f;
    if (m < Cout && kk < K) v = (_Float16)w[(long long)m * K + kk];
    dst[i] = v;
}

// ---------------------------------------------------------------------------
// Concat two [B,3,H,W] tensors along C -> [B,6,H,W]
// ---------------------------------------------------------------------------
__global__ void concat6_kernel(const float* __restrict__ a,
                               const float* __restrict__ b,
                               float* __restrict__ out, int HW) {
    int i = blockIdx.x * blockDim.x + threadIdx.x;
    int total = 2 * 6 * HW;  // B = 2
    if (i >= total) return;
    int hw = i % HW;
    int c  = (i / HW) % 6;
    int n  = i / (6 * HW);
    out[i] = (c < 3) ? a[(n * 3 + c) * HW + hw] : b[(n * 3 + (c - 3)) * HW + hw];
}

// ---------------------------------------------------------------------------
// 2x2 average pool; B folded into C
// ---------------------------------------------------------------------------
__global__ void avgpool2_kernel(const float* __restrict__ in,
                                float* __restrict__ out, int C, int H, int W) {
    int Ho = H >> 1, Wo = W >> 1;
    int i = blockIdx.x * blockDim.x + threadIdx.x;
    int total = C * Ho * Wo;
    if (i >= total) return;
    int xo = i % Wo;
    int yo = (i / Wo) % Ho;
    int c  = i / (Wo * Ho);
    const float* p = in + ((long long)c * H + 2 * yo) * W + 2 * xo;
    out[i] = 0.25f * (p[0] + p[1] + p[W] + p[W + 1]);
}

// ---------------------------------------------------------------------------
// Bilinear x2 upsample, align_corners = true; B folded into C
// ---------------------------------------------------------------------------
__global__ void up2_ac_kernel(const float* __restrict__ in,
                              float* __restrict__ out, int C, int H, int W) {
    int Ho = 2 * H, Wo = 2 * W;
    long long total = (long long)C * Ho * Wo;
    long long i = (long long)blockIdx.x * blockDim.x + threadIdx.x;
    if (i >= total) return;
    int xo = (int)(i % Wo);
    long long t = i / Wo;
    int yo = (int)(t % Ho);
    int c  = (int)(t / Ho);
    float sy = (float)(H - 1) / (float)(Ho - 1);
    float sx = (float)(W - 1) / (float)(Wo - 1);
    float fy = yo * sy, fx = xo * sx;
    int y0 = (int)fy, x0 = (int)fx;
    float ty = fy - (float)y0, tx = fx - (float)x0;
    int y1 = imin(y0 + 1, H - 1), x1 = imin(x0 + 1, W - 1);
    const float* p = in + (long long)c * H * W;
    float v00 = p[y0 * W + x0], v01 = p[y0 * W + x1];
    float v10 = p[y1 * W + x0], v11 = p[y1 * W + x1];
    out[i] = (1.f - ty) * ((1.f - tx) * v00 + tx * v01) +
             ty         * ((1.f - tx) * v10 + tx * v11);
}

// ---------------------------------------------------------------------------
// Conv3x3 (SAME, zero pad) as implicit GEMM on WMMA f32<=f16*f16.
// Workgroup (256 thr = 8 waves) computes a 64(Cout) x 64(pixel) tile.
// A fragments come straight from the pre-packed global weight buffer
// (one 32B contiguous load per lane per K-step). The im2col B panel
// ([64pix][32k] f16) is staged into double-buffered LDS with independent
// gather registers (branch-free; skip-add resolved at compile time via
// template), zero-pad via select; each wave issues 2 WMMAs per K-step
// reusing its A fragment. One barrier per K-step. The (ci,dy,dx) im2col
// decomposition is maintained incrementally -- no divisions, 32-bit math.
// ---------------------------------------------------------------------------
template <bool HAS_IN2>
__global__ __launch_bounds__(256) void conv3x3_wmma_kernel(
    const float* __restrict__ in, const float* __restrict__ in2,
    const _Float16* __restrict__ whs, const float* __restrict__ bias,
    float* __restrict__ out, int Cin, int Cout, int H, int W, int relu) {
    __shared__ __align__(64) _Float16 Bsub[2][64 * 32];  // [pix][k] ping-pong

    const int t    = threadIdx.x;
    const int lane = t & 31;
    const int wv   = t >> 5;          // wave id 0..7
    const int half = lane >> 4;       // 0 | 1
    const int l16  = lane & 15;
    const int HW   = H * W;
    const int b    = blockIdx.z;
    const int coBase  = blockIdx.y * 64;
    const int pixBase = blockIdx.x * 64;
    const int K   = Cin * 9;
    const int K32 = (K + 31) >> 5;

    const float* inb  = in + b * Cin * HW;
    const float* in2b = HAS_IN2 ? in2 + b * Cin * HW : nullptr;

    union { v8f v; float f[8]; } acc0, acc1;
    #pragma unroll
    for (int r = 0; r < 8; ++r) { acc0.f[r] = 0.f; acc1.f[r] = 0.f; }

    const int ct = wv & 3;            // co sub-tile (16 rows) within 64
    const int pp = (wv >> 2) * 32;    // base of this wave's two pixel tiles

    // Per-lane base of pre-packed A fragments for this wave's co-tile.
    const int ctg = blockIdx.y * 4 + ct;
    const _Float16* abase = whs + ((long long)ctg * K32) * 512 + lane * 16;

    // ---- per-thread staging invariants (8 im2col elements / thread) ----
    // k = 9*ci + 3*q3 + r3 ;  dy = q3-1, dx = r3-1. Advance +32 per K-step.
    int kle[8], ldso[8], ye[8], xe[8], cie[8], q3e[8], r3e[8];
    #pragma unroll
    for (int e = 0; e < 8; ++e) {
        int lin = e * 256 + t;                 // 0..2047
        int pl = lin & 63, kl = lin >> 6;
        kle[e]  = kl;
        ldso[e] = pl * 32 + kl;
        int pix = pixBase + pl;
        int pc  = pix < HW ? pix : HW - 1;
        ye[e] = pc / W;
        xe[e] = pc - ye[e] * W;
        int ci = kl / 9;                        // divisions at init only
        int r9 = kl - ci * 9;
        int q3 = r9 / 3;
        cie[e] = ci; q3e[e] = q3; r3e[e] = r9 - q3 * 3;
    }

    #pragma unroll 2
    for (int ks = 0; ks < K32; ++ks) {
        const int k0 = ks * 32;
        _Float16* bs = &Bsub[ks & 1][0];

        // ---- gather im2col values into independent registers ----
        float bvals[8], bvals2[8];
        int   bins[8];
        #pragma unroll
        for (int e = 0; e < 8; ++e) {
            int dy = q3e[e] - 1, dx = r3e[e] - 1;
            int yy = ye[e] + dy, xx = xe[e] + dx;
            int kk = k0 + kle[e];
            bins[e] = (yy >= 0) & (yy < H) & (xx >= 0) & (xx < W) & (kk < K);
            int ic  = imin(cie[e], Cin - 1);
            int idx = ic * HW + imin(imax(yy, 0), H - 1) * W +
                      imin(imax(xx, 0), W - 1);
            bvals[e] = inb[idx];
            if (HAS_IN2) bvals2[e] = in2b[idx];
        }
        #pragma unroll
        for (int e = 0; e < 8; ++e) {
            float v = bvals[e];
            if (HAS_IN2) v += bvals2[e];
            bs[ldso[e]] = bins[e] ? (_Float16)v : (_Float16)0.f;
        }
        // advance k by 32 = 9*3 + 3*1 + 2 with carries
        #pragma unroll
        for (int e = 0; e < 8; ++e) {
            r3e[e] += 2; q3e[e] += 1; cie[e] += 3;
            if (r3e[e] >= 3) { r3e[e] -= 3; q3e[e] += 1; }
            if (q3e[e] >= 3) { q3e[e] -= 3; cie[e] += 1; }
        }

        // ---- A fragment: one contiguous 32B global load per lane ----
        v16h afv = *(const v16h*)(abase + (long long)ks * 512);
        if (ks + 1 < K32)
            __builtin_prefetch((const void*)(abase + (long long)(ks + 1) * 512), 0, 1);

        __syncthreads();

        // ---- B fragments: lane n, K 0..15 (half 0) / 16..31 (half 1) ----
        const int n0 = pp + l16, n1 = pp + 16 + l16;
        v16h b0 = *(const v16h*)(bs + n0 * 32 + half * 16);
        v16h b1 = *(const v16h*)(bs + n1 * 32 + half * 16);

        acc0.v = __builtin_amdgcn_wmma_f32_16x16x32_f16(
            false, afv, false, b0, (short)0, acc0.v, false, false);
        acc1.v = __builtin_amdgcn_wmma_f32_16x16x32_f16(
            false, afv, false, b1, (short)0, acc1.v, false, false);
    }

    // ---- epilogue: bias (+ReLU); D layout: VGPR r -> M = r + 8*half ----
    #pragma unroll
    for (int r = 0; r < 8; ++r) {
        int co = coBase + ct * 16 + r + half * 8;
        if (co < Cout) {
            float bv = bias[co];
            int obase = (b * Cout + co) * HW;
            int p0 = pixBase + pp + l16;
            if (p0 < HW) {
                float v = acc0.f[r] + bv;
                if (relu) v = fmaxf(v, 0.f);
                out[obase + p0] = v;
            }
            int p1 = pixBase + pp + 16 + l16;
            if (p1 < HW) {
                float v = acc1.f[r] + bv;
                if (relu) v = fmaxf(v, 0.f);
                out[obase + p1] = v;
            }
        }
    }
}

// ---------------------------------------------------------------------------
// Adaptive separable 51x51 convolution with replicate padding.
// Block = 256 threads handling a 16x16 output tile of one (b,c) plane.
// The replicate-padded 66x66 input patch is staged into LDS with
// gfx1250 async-to-LDS loads (padding == address clamping, no select),
// then the factored 51(vert)+51(horiz) contraction runs out of LDS.
// ---------------------------------------------------------------------------
#define SK    51
#define SPAD  25
#define STILE 16
#define SLDS  (STILE + SK - 1)   // 66

__global__ __launch_bounds__(256) void sepconv51_kernel(
    const float* __restrict__ inp, const float* __restrict__ vert,
    const float* __restrict__ horiz, float* __restrict__ out,
    int H, int W, int accum) {
    __shared__ float tile[SLDS * SLDS];  // 17424 B

    const int t  = threadIdx.x;
    const int bc = blockIdx.z;            // b*3 + c
    const int b  = bc / 3;
    const int x0 = blockIdx.x * STILE;
    const int y0 = blockIdx.y * STILE;
    const long long HW = (long long)H * W;
    const float* ip = inp + (long long)bc * HW;

    // ---- stage padded patch via async loads (replicate pad = clamp) ----
    const int total = SLDS * SLDS;        // 4356
    #pragma unroll
    for (int e = 0; e < (total + 255) / 256; ++e) {
        int lin = t + e * 256;
        if (lin >= total) lin = total - 1;            // duplicated tail is harmless
        int ty = lin / SLDS, tx = lin - ty * SLDS;
        int gy = imin(imax(y0 + ty - SPAD, 0), H - 1);
        int gx = imin(imax(x0 + tx - SPAD, 0), W - 1);
        unsigned long long gaddr =
            (unsigned long long)(uintptr_t)(ip + (long long)gy * W + gx);
        unsigned ldsOff = (unsigned)(uintptr_t)(tile + lin);
        asm volatile("global_load_async_to_lds_b32 %0, %1, off"
                     :: "v"(ldsOff), "v"(gaddr) : "memory");
    }
    asm volatile("s_wait_asynccnt 0x0" ::: "memory");
    __syncthreads();

    // ---- factored separable accumulation ----
    const int lx = t & 15, ly = t >> 4;
    const int x = x0 + lx, y = y0 + ly;
    const long long vb = ((long long)b * SK * H + y) * W + x;  // +k*HW per tap

    float vk[SK];
    #pragma unroll
    for (int k = 0; k < SK; ++k) vk[k] = vert[vb + (long long)k * HW];

    float acc = 0.f;
    for (int j = 0; j < SK; ++j) {
        float s = 0.f;
        #pragma unroll
        for (int k = 0; k < SK; ++k)
            s += vk[k] * tile[(ly + k) * SLDS + (lx + j)];
        acc += horiz[vb + (long long)j * HW] * s;
    }

    long long oi = (long long)bc * HW + (long long)y * W + x;
    if (accum) out[oi] += acc; else out[oi] = acc;
}

// ---------------------------------------------------------------------------
// Host orchestration
// ---------------------------------------------------------------------------
static inline int cdiv(int a, int b) { return (a + b - 1) / b; }

extern "C" void kernel_launch(void* const* d_in, const int* in_sizes, int n_in,
                              void* d_out, int out_size, void* d_ws, size_t ws_size,
                              hipStream_t stream) {
    (void)in_sizes; (void)n_in; (void)out_size; (void)ws_size;

    // (cin, cout) per weight pair, in _SPEC order.
    static const int SPEC[47][2] = {
        {6,32},{32,32},{32,32},            // conv1   0-2
        {32,64},{64,64},{64,64},           // conv2   3-5
        {64,128},{128,128},{128,128},      // conv3   6-8
        {128,256},{256,256},{256,256},     // conv4   9-11
        {256,512},{512,512},{512,512},     // conv5   12-14
        {512,512},{512,512},{512,512},     // deconv5 15-17
        {512,256},{256,256},{256,256},     // deconv4 18-20
        {256,128},{128,128},{128,128},     // deconv3 21-23
        {128,64},{64,64},{64,64},          // deconv2 24-26
        {512,512},{256,256},{128,128},{64,64},  // up5..up2 27-30
        {64,64},{64,64},{64,51},{51,51},   // vert1   31-34
        {64,64},{64,64},{64,51},{51,51},   // vert2   35-38
        {64,64},{64,64},{64,51},{51,51},   // horiz1  39-42
        {64,64},{64,64},{64,51},{51,51},   // horiz2  43-46
    };

    const float* tFirst  = (const float*)d_in[0];
    const float* tSecond = (const float*)d_in[1];
    const float* Wp[47];
    const float* Bp[47];
    for (int i = 0; i < 47; ++i) {
        Wp[i] = (const float*)d_in[2 + 2 * i];
        Bp[i] = (const float*)d_in[3 + 2 * i];
    }

    // Bump allocator over the workspace.
    char*  base = (char*)d_ws;
    size_t off  = 0;
    auto alloc = [&](size_t bytes) -> void* {
        size_t a = (off + 255) & ~(size_t)255;
        void* p = base + a;
        off = a + bytes;
        return p;
    };

    // Pre-packed f16 weight fragments in workspace.
    _Float16* Wh[47];
    for (int i = 0; i < 47; ++i) {
        int Cin = SPEC[i][0], Cout = SPEC[i][1];
        int K   = Cin * 9;
        int K32 = cdiv(K, 32);
        int CoT = cdiv(Cout, 64) * 4;          // co-tiles covered by the grid
        long long n = (long long)CoT * K32 * 512;
        Wh[i] = (_Float16*)alloc((size_t)n * sizeof(_Float16));
        pack_wmma_weights_kernel<<<(unsigned)((n + 255) / 256), 256, 0, stream>>>(
            Wp[i], Wh[i], Cout, K, K32, CoT);
    }

    const int B = 2;
    const size_t ABUF = (size_t)B * 51 * 256 * 256;  // covers every intermediate
    float* x0 = (float*)alloc((size_t)B * 6 * 256 * 256 * sizeof(float));
    float* A  = (float*)alloc(ABUF * sizeof(float));
    float* Bb = (float*)alloc(ABUF * sizeof(float));
    float* c2 = (float*)alloc((size_t)B * 64  * 128 * 128 * sizeof(float));
    float* c3 = (float*)alloc((size_t)B * 128 *  64 *  64 * sizeof(float));
    float* c4 = (float*)alloc((size_t)B * 256 *  32 *  32 * sizeof(float));
    float* c5 = (float*)alloc((size_t)B * 512 *  16 *  16 * sizeof(float));
    float* d5 = (float*)alloc((size_t)B * 512 *  16 *  16 * sizeof(float));
    float* d4 = (float*)alloc((size_t)B * 256 *  32 *  32 * sizeof(float));
    float* d3 = (float*)alloc((size_t)B * 128 *  64 *  64 * sizeof(float));
    float* d2 = (float*)alloc((size_t)B * 64  * 128 * 128 * sizeof(float));
    float* vb = (float*)alloc((size_t)B * 51  * 256 * 256 * sizeof(float));
    float* hb = (float*)alloc((size_t)B * 51  * 256 * 256 * sizeof(float));

    auto conv = [&](const float* in, const float* in2, int pi, float* out,
                    int H, int W, int relu) {
        int Cin = SPEC[pi][0], Cout = SPEC[pi][1];
        dim3 grid(cdiv(H * W, 64), cdiv(Cout, 64), B);
        if (in2)
            conv3x3_wmma_kernel<true><<<grid, 256, 0, stream>>>(
                in, in2, Wh[pi], Bp[pi], out, Cin, Cout, H, W, relu);
        else
            conv3x3_wmma_kernel<false><<<grid, 256, 0, stream>>>(
                in, nullptr, Wh[pi], Bp[pi], out, Cin, Cout, H, W, relu);
    };
    auto pool = [&](const float* in, float* out, int C, int H, int W) {
        int total = C * (H / 2) * (W / 2);
        avgpool2_kernel<<<cdiv(total, 256), 256, 0, stream>>>(in, out, C, H, W);
    };
    auto up2 = [&](const float* in, float* out, int C, int H, int W) {
        long long total = (long long)C * 2 * H * 2 * W;
        up2_ac_kernel<<<(unsigned)((total + 255) / 256), 256, 0, stream>>>(in, out, C, H, W);
    };

    // ---- Encoder ----
    concat6_kernel<<<cdiv(2 * 6 * 256 * 256, 256), 256, 0, stream>>>(tFirst, tSecond, x0, 256 * 256);
    conv(x0, nullptr, 0, A, 256, 256, 1);
    conv(A,  nullptr, 1, Bb, 256, 256, 1);
    conv(Bb, nullptr, 2, A, 256, 256, 1);                 // c1 = A
    pool(A, Bb, B * 32, 256, 256);                        // p1 = Bb
    conv(Bb, nullptr, 3, A, 128, 128, 1);
    conv(A,  nullptr, 4, Bb, 128, 128, 1);
    conv(Bb, nullptr, 5, c2, 128, 128, 1);                // c2 kept
    pool(c2, A, B * 64, 128, 128);                        // p2 = A
    conv(A,  nullptr, 6, Bb, 64, 64, 1);
    conv(Bb, nullptr, 7, A, 64, 64, 1);
    conv(A,  nullptr, 8, c3, 64, 64, 1);                  // c3 kept
    pool(c3, A, B * 128, 64, 64);                         // p3 = A
    conv(A,  nullptr, 9, Bb, 32, 32, 1);
    conv(Bb, nullptr, 10, A, 32, 32, 1);
    conv(A,  nullptr, 11, c4, 32, 32, 1);                 // c4 kept
    pool(c4, A, B * 256, 32, 32);                         // p4 = A
    conv(A,  nullptr, 12, Bb, 16, 16, 1);
    conv(Bb, nullptr, 13, A, 16, 16, 1);
    conv(A,  nullptr, 14, c5, 16, 16, 1);                 // c5 kept
    pool(c5, A, B * 512, 16, 16);                         // p5 = A (8x8)

    // ---- Decoder ----
    conv(A,  nullptr, 15, Bb, 8, 8, 1);
    conv(Bb, nullptr, 16, A, 8, 8, 1);
    conv(A,  nullptr, 17, Bb, 8, 8, 1);                   // deconv5 = Bb
    up2(Bb, A, B * 512, 8, 8);
    conv(A,  nullptr, 27, d5, 16, 16, 1);                 // up5 -> d5

    conv(d5, c5, 18, A, 16, 16, 1);                       // _basic(d5+c5)
    conv(A,  nullptr, 19, Bb, 16, 16, 1);
    conv(Bb, nullptr, 20, A, 16, 16, 1);
    up2(A, Bb, B * 256, 16, 16);
    conv(Bb, nullptr, 28, d4, 32, 32, 1);                 // up4 -> d4

    conv(d4, c4, 21, A, 32, 32, 1);
    conv(A,  nullptr, 22, Bb, 32, 32, 1);
    conv(Bb, nullptr, 23, A, 32, 32, 1);
    up2(A, Bb, B * 128, 32, 32);
    conv(Bb, nullptr, 29, d3, 64, 64, 1);                 // up3 -> d3

    conv(d3, c3, 24, A, 64, 64, 1);
    conv(A,  nullptr, 25, Bb, 64, 64, 1);
    conv(Bb, nullptr, 26, A, 64, 64, 1);
    up2(A, Bb, B * 64, 64, 64);
    conv(Bb, nullptr, 30, d2, 128, 128, 1);               // up2 -> d2

    // ---- Kernel-prediction subnets (comb = d2 + c2 fused as in2) ----
    auto subnet = [&](int p0, float* dst) {
        conv(d2, c2, p0 + 0, A, 128, 128, 1);
        conv(A,  nullptr, p0 + 1, Bb, 128, 128, 1);
        conv(Bb, nullptr, p0 + 2, A, 128, 128, 1);        // 51ch @ 128^2
        up2(A, Bb, B * 51, 128, 128);                     // 51ch @ 256^2
        conv(Bb, nullptr, p0 + 3, dst, 256, 256, 0);      // no relu
    };

    float* outp = (float*)d_out;
    dim3 sgrid(256 / STILE, 256 / STILE, 2 * 3);

    subnet(31, vb);  // vert1
    subnet(39, hb);  // horiz1
    sepconv51_kernel<<<sgrid, 256, 0, stream>>>(tFirst, vb, hb, outp, 256, 256, 0);

    subnet(35, vb);  // vert2
    subnet(43, hb);  // horiz2
    sepconv51_kernel<<<sgrid, 256, 0, stream>>>(tSecond, vb, hb, outp, 256, 256, 1);
}
